// GraphAttentionLayer_2_65343632441420
// MI455X (gfx1250) — matrix operand order repaired and checked by reading, not compiled, not run on hardware
//
#include <hip/hip_runtime.h>

typedef __attribute__((ext_vector_type(16))) _Float16 v16h;
typedef __attribute__((ext_vector_type(8)))  _Float16 v8h;
typedef __attribute__((ext_vector_type(8)))  float    v8f;

#define N_NODES 8192
#define F_DIM   64
#define NW      8            // waves per block in the attention kernel
#define NEG_BIG (-9.0e15f)

// ---------------------------------------------------------------------------
// Kernel 1: Wh = X @ W (fp32), store WhT (f16, transposed [64][8192]),
//           src = Wh @ a[:64], dst = Wh @ a[64:]
// One block of 64 threads per node row.
// ---------------------------------------------------------------------------
__global__ __launch_bounds__(64) void wh_prep_kernel(
    const float* __restrict__ X, const float* __restrict__ W,
    const float* __restrict__ a,
    _Float16* __restrict__ WhT, float* __restrict__ srcv, float* __restrict__ dstv)
{
  __shared__ float xs[64];
  __shared__ float red[64];
  const int row = blockIdx.x;
  const int t   = threadIdx.x;

  xs[t] = X[row * F_DIM + t];
  __syncthreads();

  float acc = 0.f;
#pragma unroll
  for (int k = 0; k < F_DIM; ++k) acc = fmaf(xs[k], W[k * F_DIM + t], acc);

  WhT[(size_t)t * N_NODES + row] = (_Float16)acc;

  red[t] = acc * a[t];
  __syncthreads();
  for (int s = 32; s > 0; s >>= 1) {
    if (t < s) red[t] += red[t + s];
    __syncthreads();
  }
  if (t == 0) srcv[row] = red[0];
  __syncthreads();

  red[t] = acc * a[F_DIM + t];
  __syncthreads();
  for (int s = 32; s > 0; s >>= 1) {
    if (t < s) red[t] += red[t + s];
    __syncthreads();
  }
  if (t == 0) dstv[row] = red[0];
}

// ---------------------------------------------------------------------------
// Kernel 2: fused masked-softmax attention + (attention @ Wh) + ELU.
// Block = 16 output rows, NW waves split the 8192-wide j dimension.
// Flash (online-softmax) accumulation through v_wmma_f32_16x16x32_f16.
// ---------------------------------------------------------------------------
__global__ __launch_bounds__(32 * NW) void gat_attn_kernel(
    const int* __restrict__ adj, const float* __restrict__ srcv,
    const float* __restrict__ dstv, const _Float16* __restrict__ WhT,
    float* __restrict__ out)
{
  __shared__ float sh_m[NW][16];
  __shared__ float sh_l[NW][16];
  __shared__ float sh_O[NW][16][F_DIM];

  const int tid    = threadIdx.x;
  const int lane   = tid & 31;
  const int wave   = tid >> 5;
  const int i0     = blockIdx.x * 16;
  const int row16  = lane & 15;     // row within tile (A-matrix M, B-matrix N)
  const int hiHalf = lane >> 4;     // 0: K {0-7,16-23}; 1: K {8-15,24-31}
  const int kb0    = hiHalf * 8;

  const float  s_i     = srcv[i0 + row16];
  const size_t rowbase = (size_t)(i0 + row16) * N_NODES;

  float m    = -INFINITY;
  float lsum = 0.f;
  v8f acc0 = {}, acc1 = {}, acc2 = {}, acc3 = {};

  for (int j0 = wave * 32; j0 < N_NODES; j0 += 32 * NW) {
    const int base0 = j0 + kb0;
    // adjacency (16 ints/lane, matching A-fragment K order)
    const int4* ap0 = (const int4*)(adj + rowbase + base0);
    const int4* ap1 = (const int4*)(adj + rowbase + base0 + 16);
    int4 A0 = ap0[0], A1 = ap0[1], A2 = ap1[0], A3 = ap1[1];
    // dst scores for the same K set (L2-resident, 32 KB total)
    const float4* dp0 = (const float4*)(dstv + base0);
    const float4* dp1 = (const float4*)(dstv + base0 + 16);
    float4 D0 = dp0[0], D1 = dp0[1], D2 = dp1[0], D3 = dp1[1];

    auto mke = [&](int av, float dv) -> float {
      float e  = s_i + dv;
      float le = fmaxf(e, 0.2f * e);       // LeakyReLU(0.2)
      return (av > 0) ? le : NEG_BIG;      // adjacency mask
    };

    float ev[16];
    ev[ 0] = mke(A0.x, D0.x); ev[ 1] = mke(A0.y, D0.y);
    ev[ 2] = mke(A0.z, D0.z); ev[ 3] = mke(A0.w, D0.w);
    ev[ 4] = mke(A1.x, D1.x); ev[ 5] = mke(A1.y, D1.y);
    ev[ 6] = mke(A1.z, D1.z); ev[ 7] = mke(A1.w, D1.w);
    ev[ 8] = mke(A2.x, D2.x); ev[ 9] = mke(A2.y, D2.y);
    ev[10] = mke(A2.z, D2.z); ev[11] = mke(A2.w, D2.w);
    ev[12] = mke(A3.x, D3.x); ev[13] = mke(A3.y, D3.y);
    ev[14] = mke(A3.z, D3.z); ev[15] = mke(A3.w, D3.w);

    // per-row tile max (row split across lane pair l / l^16)
    float tmax = ev[0];
#pragma unroll
    for (int k = 1; k < 16; ++k) tmax = fmaxf(tmax, ev[k]);
    tmax = fmaxf(tmax, __shfl_xor(tmax, 16, 32));
    const float mnew = fmaxf(m, tmax);

    // rescale accumulators only when some row's max actually moved
    if (__any(mnew > m)) {
      const float alpha = __expf(m - mnew);
      lsum *= alpha;
      const int rb = hiHalf * 8;           // C-layout rows this lane holds
      float af[8];
#pragma unroll
      for (int r = 0; r < 8; ++r) af[r] = __shfl(alpha, rb + r, 32);
#pragma unroll
      for (int r = 0; r < 8; ++r) {
        acc0[r] *= af[r]; acc1[r] *= af[r];
        acc2[r] *= af[r]; acc3[r] *= af[r];
      }
    }
    m = mnew;

    // probabilities -> f16 A fragment (16 values/lane in A-layout order)
    v16h Pa;
    float ps = 0.f;
#pragma unroll
    for (int k = 0; k < 16; ++k) {
      float p = __expf(ev[k] - m);
      ps += p;
      Pa[k] = (_Float16)p;
    }
    lsum += ps;

    // B fragments: WhT[f][j], lane = column f (mod 16), K {8h..8h+7, 16+8h..16+8h+7} per lane
    const _Float16* bb = WhT + (size_t)(j0 + hiHalf * 8);
    auto ldb = [&](const _Float16* p) -> v16h {
      v8h lo = *(const v8h*)p, hi = *(const v8h*)(p + 16);
      return __builtin_shufflevector(lo, hi, 0,1,2,3,4,5,6,7,8,9,10,11,12,13,14,15);
    };
    v16h B0 = ldb(bb + (size_t)(row16     ) * N_NODES);
    v16h B1 = ldb(bb + (size_t)(row16 + 16) * N_NODES);
    v16h B2 = ldb(bb + (size_t)(row16 + 32) * N_NODES);
    v16h B3 = ldb(bb + (size_t)(row16 + 48) * N_NODES);

    acc0 = __builtin_amdgcn_wmma_f32_16x16x32_f16(false, Pa, false, B0, (short)0, acc0, false, false);
    acc1 = __builtin_amdgcn_wmma_f32_16x16x32_f16(false, Pa, false, B1, (short)0, acc1, false, false);
    acc2 = __builtin_amdgcn_wmma_f32_16x16x32_f16(false, Pa, false, B2, (short)0, acc2, false, false);
    acc3 = __builtin_amdgcn_wmma_f32_16x16x32_f16(false, Pa, false, B3, (short)0, acc3, false, false);
  }

  // publish per-wave partials
  const float ltot = lsum + __shfl_xor(lsum, 16, 32);
  if (lane < 16) { sh_m[wave][lane] = m; sh_l[wave][lane] = ltot; }
#pragma unroll
  for (int r = 0; r < 8; ++r) {
    const int row = r + hiHalf * 8;        // C/D layout: M = r (+8 for hi lanes)
    sh_O[wave][row][row16 +  0] = acc0[r];
    sh_O[wave][row][row16 + 16] = acc1[r];
    sh_O[wave][row][row16 + 32] = acc2[r];
    sh_O[wave][row][row16 + 48] = acc3[r];
  }
  __syncthreads();

  // merge NW partials (flash-decoding combine), normalize, ELU, store.
  // 256 threads over 16x64 outputs -> 4 columns each.
  const int row = tid >> 4;                // 0..15
  const int c0  = (tid & 15) * 4;          // 0..60
  float mf = sh_m[0][row];
#pragma unroll
  for (int w = 1; w < NW; ++w) mf = fmaxf(mf, sh_m[w][row]);
  float wgt[NW];
  float L = 0.f;
#pragma unroll
  for (int w = 0; w < NW; ++w) {
    wgt[w] = __expf(sh_m[w][row] - mf);
    L += wgt[w] * sh_l[w][row];
  }
  const float invL = 1.0f / L;
#pragma unroll
  for (int c = 0; c < 4; ++c) {
    const int cc = c0 + c;
    float o = 0.f;
#pragma unroll
    for (int w = 0; w < NW; ++w) o += wgt[w] * sh_O[w][row][cc];
    const float h = o * invL;
    out[(size_t)(i0 + row) * F_DIM + cc] = (h > 0.f) ? h : (__expf(h) - 1.f);
  }
}

// ---------------------------------------------------------------------------
extern "C" void kernel_launch(void* const* d_in, const int* in_sizes, int n_in,
                              void* d_out, int out_size, void* d_ws, size_t ws_size,
                              hipStream_t stream) {
  const float* X   = (const float*)d_in[0];   // (8192, 64) f32
  const int*   adj = (const int*)  d_in[1];   // (8192, 8192) i32
  const float* W   = (const float*)d_in[2];   // (64, 64) f32
  const float* a   = (const float*)d_in[3];   // (128, 1) f32
  float* out = (float*)d_out;                 // (8192, 64) f32

  char* ws = (char*)d_ws;
  _Float16* WhT  = (_Float16*)ws;                                  // 1 MB
  float*    srcv = (float*)(ws + (size_t)N_NODES * F_DIM * sizeof(_Float16));
  float*    dstv = srcv + N_NODES;                                 // 32 KB each

  wh_prep_kernel<<<N_NODES, 64, 0, stream>>>(X, W, a, WhT, srcv, dstv);
  gat_attn_kernel<<<N_NODES / 16, 32 * NW, 0, stream>>>(adj, srcv, dstv, WhT, out);
}
